// CausalSelfAttention_21096879358080
// MI455X (gfx1250) — matrix operand order repaired and checked by reading, not compiled, not run on hardware
//
#include <hip/hip_runtime.h>

typedef __attribute__((ext_vector_type(16))) _Float16 v16h;
typedef __attribute__((ext_vector_type(8)))  _Float16 v8h;
typedef __attribute__((ext_vector_type(8)))  float    v8f;

constexpr int Bn = 8, Tn = 2048, Cn = 128, Hn = 4, Dn = 32;
constexpr int MTOT = Bn * Tn;            // 16384 rows over (b,t)
constexpr float QSCALE = 0.17677669529663687f;  // 1/sqrt(32)

__device__ __forceinline__ v8f vzero8() {
  v8f z;
#pragma unroll
  for (int i = 0; i < 8; ++i) z[i] = 0.0f;
  return z;
}

// Build a 16-half fragment from two contiguous 16B chunks.
__device__ __forceinline__ v16h frag16(const _Float16* p0, const _Float16* p1) {
  v8h a = *(const v8h*)p0;
  v8h b = *(const v8h*)p1;
  v16h r;
#pragma unroll
  for (int i = 0; i < 8; ++i) { r[i] = a[i]; r[i + 8] = b[i]; }
  return r;
}

__device__ __forceinline__ float rowmax16(float x) {
#pragma unroll
  for (int m = 1; m < 16; m <<= 1) x = fmaxf(x, __shfl_xor(x, m, 32));
  return x;
}
__device__ __forceinline__ float rowsum16(float x) {
#pragma unroll
  for (int m = 1; m < 16; m <<= 1) x += __shfl_xor(x, m, 32);
  return x;
}

// ---------------- conversion kernels ----------------
__global__ void cvt_x_kernel(const float* __restrict__ x, _Float16* __restrict__ xh, int n) {
  int i = blockIdx.x * blockDim.x + threadIdx.x;
  if (i < n) xh[i] = (_Float16)x[i];
}

// wT[n][k] = w[k][n]  (f16, so B-fragments load 16 consecutive k per lane)
__global__ void cvt_w_kernel(const float* __restrict__ wqkv, const float* __restrict__ wproj,
                             _Float16* __restrict__ wqkvT, _Float16* __restrict__ wprojT) {
  int i = blockIdx.x * blockDim.x + threadIdx.x;
  if (i < 384 * 128) {
    int n = i >> 7, k = i & 127;
    wqkvT[i] = (_Float16)wqkv[k * 384 + n];
  }
  int j = i - 384 * 128;
  if (j >= 0 && j < 128 * 128) {
    int n = j >> 7, k = j & 127;
    wprojT[j] = (_Float16)wproj[k * 128 + n];
  }
}

// ---------------- QKV projection: [16384,128] @ [128,384] + b ----------------
// One 16x64 output strip per wave (4 n-subtiles reuse each A fragment).
__global__ void __launch_bounds__(256) gemm_qkv_kernel(const _Float16* __restrict__ xh,
                                                       const _Float16* __restrict__ wT,
                                                       const float* __restrict__ bias,
                                                       _Float16* __restrict__ Qh,
                                                       _Float16* __restrict__ Kh,
                                                       _Float16* __restrict__ Vt) {
  constexpr int NG = 384 / 64;  // 6 n-groups of 64
  int wt = blockIdx.x * (blockDim.x >> 5) + (threadIdx.x >> 5);
  if (wt >= (MTOT / 16) * NG) return;
  int lane = threadIdx.x & 31;
  int half = lane >> 4, l16 = lane & 15;
  int m0 = (wt / NG) * 16, n0 = (wt % NG) * 64;

  v8f acc[4];
#pragma unroll
  for (int s = 0; s < 4; ++s) {
    float bv = bias[n0 + s * 16 + l16];
#pragma unroll
    for (int i = 0; i < 8; ++i) acc[s][i] = bv;  // C/D: lane = col n, bias per-col
  }

  const _Float16* arow = xh + (size_t)(m0 + l16) * Cn;
  const _Float16* brow[4];
#pragma unroll
  for (int s = 0; s < 4; ++s) brow[s] = wT + (size_t)(n0 + s * 16 + l16) * Cn;

#pragma unroll
  for (int k0 = 0; k0 < Cn; k0 += 32) {
    v16h af = frag16(arow + k0 + 16 * half, arow + k0 + 16 * half + 8);
#pragma unroll
    for (int s = 0; s < 4; ++s) {
      v16h bf = frag16(brow[s] + k0 + 16 * half, brow[s] + k0 + 16 * half + 8);
      acc[s] = __builtin_amdgcn_wmma_f32_16x16x32_f16(false, af, false, bf, (short)0, acc[s], false, false);
    }
  }

#pragma unroll
  for (int s = 0; s < 4; ++s) {
    int n = n0 + s * 16 + l16;
    int which = n >> 7;  // 0=q, 1=k, 2=v
    int h = (n >> 5) & 3;
    int d = n & 31;
#pragma unroll
    for (int v = 0; v < 8; ++v) {
      int m = m0 + half * 8 + v;  // flat (b,t)
      int b = m >> 11, t = m & (Tn - 1);
      int bh = b * Hn + h;
      float val = acc[s][v];
      if (which == 0)
        Qh[(size_t)(bh * Tn + t) * Dn + d] = (_Float16)(val * QSCALE);
      else if (which == 1)
        Kh[(size_t)(bh * Tn + t) * Dn + d] = (_Float16)val;
      else
        Vt[(size_t)(bh * Dn + d) * Tn + t] = (_Float16)val;
    }
  }
}

// ---------------- flash attention ----------------
// Block = 8 waves = 128 query rows of one (b,h). K/V 32-key blocks are staged
// into LDS once per block via async global->LDS copies (double buffered), so
// the 8 waves share them instead of re-reading global 8x.
__global__ void __launch_bounds__(256) flash_attn_kernel(const _Float16* __restrict__ Qh,
                                                         const _Float16* __restrict__ Kh,
                                                         const _Float16* __restrict__ Vt,
                                                         _Float16* __restrict__ yh) {
  __shared__ __align__(16) _Float16 kbuf[2][32][32];  // [buf][t_local][d]
  __shared__ __align__(16) _Float16 vbuf[2][32][32];  // [buf][d][t_local]
  __shared__ __align__(16) _Float16 pbuf[8][16][32];  // per-wave P staging

  int wave = threadIdx.x >> 5;
  int lane = threadIdx.x & 31;
  int half = lane >> 4, l16 = lane & 15;
  int h = blockIdx.y, b = blockIdx.z;
  int t0 = (blockIdx.x * 8 + wave) * 16;
  int bh = b * Hn + h;
  const _Float16* Q = Qh + (size_t)bh * Tn * Dn;
  const _Float16* K = Kh + (size_t)bh * Tn * Dn;
  const _Float16* V = Vt + (size_t)bh * Dn * Tn;

  const _Float16* qrow = Q + (size_t)(t0 + l16) * Dn;
  v16h qf = frag16(qrow + 16 * half, qrow + 16 * half + 8);  // K-dim = full D = 32

  float mrow[8], lrow[8];
  v8f y0 = vzero8(), y1 = vzero8();
#pragma unroll
  for (int v = 0; v < 8; ++v) { mrow[v] = -1e30f; lrow[v] = 0.0f; }

  const int kend_w = t0 + 16;                    // this wave's causal key bound
  const int blockKend = blockIdx.x * 128 + 128;  // max bound over block's waves

  // Per-lane async-copy geometry: lane covers row (lane>>2), 8 halves at col (lane&3)*8.
  int rsub = lane >> 2;
  int csub = (lane & 3) * 8;

  auto issueK = [&](int j, int buf) {
#pragma unroll
    for (int i = 0; i < 4; ++i) {
      int t = rsub + 8 * i;
      unsigned int ldso = (unsigned int)(size_t)&kbuf[buf][t][csub];
      unsigned long long ga = (unsigned long long)(size_t)(K + (size_t)(j + t) * Dn + csub);
      asm volatile("global_load_async_to_lds_b128 %0, %1, off" :: "v"(ldso), "v"(ga) : "memory");
    }
  };
  auto issueV = [&](int j, int buf) {
#pragma unroll
    for (int i = 0; i < 4; ++i) {
      int d = rsub + 8 * i;
      unsigned int ldso = (unsigned int)(size_t)&vbuf[buf][d][csub];
      unsigned long long ga = (unsigned long long)(size_t)(V + (size_t)d * Tn + j + csub);
      asm volatile("global_load_async_to_lds_b128 %0, %1, off" :: "v"(ldso), "v"(ga) : "memory");
    }
  };

  // Prologue: fill buffer 0.
  if (wave == 0) issueK(0, 0);
  else if (wave == 1) issueV(0, 0);
  if (wave < 2) asm volatile("s_wait_asynccnt 0x0" ::: "memory");
  __syncthreads();

  for (int j = 0; j < blockKend; j += 32) {
    int cur = (j >> 5) & 1;
    // Prefetch next key block into the other buffer while we compute.
    if (j + 32 < blockKend) {
      if (wave == 0) issueK(j + 32, cur ^ 1);
      else if (wave == 1) issueV(j + 32, cur ^ 1);
    }

    if (j < kend_w) {  // wave-uniform predicate; EXEC stays full for WMMA
      // S = Q * K^T (two 16-key n-tiles) from LDS
      const _Float16* kr0 = &kbuf[cur][l16][0];
      const _Float16* kr1 = &kbuf[cur][16 + l16][0];
      v16h kf0 = frag16(kr0 + 16 * half, kr0 + 16 * half + 8);
      v16h kf1 = frag16(kr1 + 16 * half, kr1 + 16 * half + 8);
      v8f s0 = __builtin_amdgcn_wmma_f32_16x16x32_f16(false, qf, false, kf0, (short)0, vzero8(), false, false);
      v8f s1 = __builtin_amdgcn_wmma_f32_16x16x32_f16(false, qf, false, kf1, (short)0, vzero8(), false, false);

      if (j + 31 > t0) {  // block touches the diagonal -> mask
        int n0g = j + l16, n1g = j + 16 + l16;
#pragma unroll
        for (int v = 0; v < 8; ++v) {
          int mg = t0 + half * 8 + v;
          if (n0g > mg) s0[v] = -1e30f;
          if (n1g > mg) s1[v] = -1e30f;
        }
      }

      float p0[8], p1[8], corr[8];
#pragma unroll
      for (int v = 0; v < 8; ++v) {
        float rm = rowmax16(fmaxf(s0[v], s1[v]));
        float mn = fmaxf(mrow[v], rm);
        corr[v] = __expf(mrow[v] - mn);
        mrow[v] = mn;
        p0[v] = __expf(s0[v] - mn);
        p1[v] = __expf(s1[v] - mn);
        lrow[v] = lrow[v] * corr[v] + rowsum16(p0[v] + p1[v]);
      }

      // stage P (D-frag: lane=col) -> LDS row-major -> reload as A-frag (lane=row)
#pragma unroll
      for (int v = 0; v < 8; ++v) {
        pbuf[wave][half * 8 + v][l16] = (_Float16)p0[v];
        pbuf[wave][half * 8 + v][16 + l16] = (_Float16)p1[v];
      }
      const _Float16* prow = &pbuf[wave][l16][0];
      v16h pf = frag16(prow + 16 * half, prow + 16 * half + 8);

#pragma unroll
      for (int v = 0; v < 8; ++v) { y0[v] *= corr[v]; y1[v] *= corr[v]; }

      const _Float16* vr0 = &vbuf[cur][l16][0];       // d-tile 0: d = l16
      const _Float16* vr1 = &vbuf[cur][16 + l16][0];  // d-tile 1
      v16h vf0 = frag16(vr0 + 16 * half, vr0 + 16 * half + 8);
      v16h vf1 = frag16(vr1 + 16 * half, vr1 + 16 * half + 8);
      y0 = __builtin_amdgcn_wmma_f32_16x16x32_f16(false, pf, false, vf0, (short)0, y0, false, false);
      y1 = __builtin_amdgcn_wmma_f32_16x16x32_f16(false, pf, false, vf1, (short)0, y1, false, false);
    }

    // Publish prefetched buffer; protect current buffer until all waves done.
    if (wave < 2) asm volatile("s_wait_asynccnt 0x0" ::: "memory");
    __syncthreads();
  }

#pragma unroll
  for (int v = 0; v < 8; ++v) {
    int t = t0 + half * 8 + v;
    float inv = 1.0f / lrow[v];
    size_t base = (size_t)(b * Tn + t) * Cn + h * Dn;
    yh[base + l16] = (_Float16)(y0[v] * inv);
    yh[base + 16 + l16] = (_Float16)(y1[v] * inv);
  }
}

// ---------------- output projection: [16384,128] @ [128,128] + b -> fp32 out ----------------
__global__ void __launch_bounds__(256) gemm_proj_kernel(const _Float16* __restrict__ yh,
                                                        const _Float16* __restrict__ wT,
                                                        const float* __restrict__ bias,
                                                        float* __restrict__ out) {
  constexpr int NG = 128 / 64;  // 2 n-groups of 64
  int wt = blockIdx.x * (blockDim.x >> 5) + (threadIdx.x >> 5);
  if (wt >= (MTOT / 16) * NG) return;
  int lane = threadIdx.x & 31;
  int half = lane >> 4, l16 = lane & 15;
  int m0 = (wt / NG) * 16, n0 = (wt % NG) * 64;

  v8f acc[4];
#pragma unroll
  for (int s = 0; s < 4; ++s) {
    float bv = bias[n0 + s * 16 + l16];
#pragma unroll
    for (int i = 0; i < 8; ++i) acc[s][i] = bv;
  }

  const _Float16* arow = yh + (size_t)(m0 + l16) * Cn;
  const _Float16* brow[4];
#pragma unroll
  for (int s = 0; s < 4; ++s) brow[s] = wT + (size_t)(n0 + s * 16 + l16) * Cn;

#pragma unroll
  for (int k0 = 0; k0 < Cn; k0 += 32) {
    v16h af = frag16(arow + k0 + 16 * half, arow + k0 + 16 * half + 8);
#pragma unroll
    for (int s = 0; s < 4; ++s) {
      v16h bf = frag16(brow[s] + k0 + 16 * half, brow[s] + k0 + 16 * half + 8);
      acc[s] = __builtin_amdgcn_wmma_f32_16x16x32_f16(false, af, false, bf, (short)0, acc[s], false, false);
    }
  }

#pragma unroll
  for (int s = 0; s < 4; ++s)
#pragma unroll
    for (int v = 0; v < 8; ++v)
      out[(size_t)(m0 + half * 8 + v) * Cn + n0 + s * 16 + l16] = acc[s][v];
}

// ---------------- host launcher ----------------
extern "C" void kernel_launch(void* const* d_in, const int* in_sizes, int n_in,
                              void* d_out, int out_size, void* d_ws, size_t ws_size,
                              hipStream_t stream) {
  const float* x      = (const float*)d_in[0];
  const float* w_qkv  = (const float*)d_in[1];
  const float* b_qkv  = (const float*)d_in[2];
  const float* w_proj = (const float*)d_in[3];
  const float* b_proj = (const float*)d_in[4];
  float* out = (float*)d_out;

  uint8_t* ws = (uint8_t*)d_ws;
  const size_t MB4 = 4u << 20;  // each f16 tensor below is exactly 4 MiB
  _Float16* xh     = (_Float16*)(ws + 0 * MB4);
  _Float16* Qh     = (_Float16*)(ws + 1 * MB4);
  _Float16* Kh     = (_Float16*)(ws + 2 * MB4);
  _Float16* Vt     = (_Float16*)(ws + 3 * MB4);
  _Float16* yh     = (_Float16*)(ws + 4 * MB4);
  _Float16* wqkvT  = (_Float16*)(ws + 5 * MB4);
  _Float16* wprojT = (_Float16*)(ws + 5 * MB4 + 384 * 128 * sizeof(_Float16));

  // 1) f16 conversions
  cvt_x_kernel<<<(MTOT * Cn) / 256, 256, 0, stream>>>(x, xh, MTOT * Cn);
  cvt_w_kernel<<<(384 * 128 + 128 * 128) / 256, 256, 0, stream>>>(w_qkv, w_proj, wqkvT, wprojT);

  // 2) QKV projection -> Q (scaled), K, V^T
  {
    int waveTiles = (MTOT / 16) * (384 / 64);
    gemm_qkv_kernel<<<waveTiles / 8, 256, 0, stream>>>(xh, wqkvT, b_qkv, Qh, Kh, Vt);
  }

  // 3) flash attention: grid = (T/128, H, B), 8 waves/block, wave = 16 query rows
  flash_attn_kernel<<<dim3(Tn / 128, Hn, Bn), 256, 0, stream>>>(Qh, Kh, Vt, yh);

  // 4) output projection -> fp32
  {
    int waveTiles = (MTOT / 16) * (128 / 64);
    gemm_proj_kernel<<<waveTiles / 8, 256, 0, stream>>>(yh, wprojT, b_proj, out);
  }
}
